// GNNGlobal_81690277970348
// MI455X (gfx1250) — matrix operand order repaired and checked by reading, not compiled, not run on hardware
//
#include <hip/hip_runtime.h>
#include <hip/hip_bf16.h>

#define NN 50000
#define EE 500000
#define DD 128
#define EDD 16
#define BB 256
#define LL 4
// Z = 2*DD + EDD = 272

typedef __attribute__((ext_vector_type(16))) _Float16 v16h;
typedef __attribute__((ext_vector_type(8)))  _Float16 v8h;
typedef __attribute__((ext_vector_type(8)))  float    v8f;

__device__ __forceinline__ void load4(const float* p, float* v) {
    float4 t = *(const float4*)p;
    v[0] = t.x; v[1] = t.y; v[2] = t.z; v[3] = t.w;
}
__device__ __forceinline__ void store4(float* p, const float* v) {
    float4 t; t.x = v[0]; t.y = v[1]; t.z = v[2]; t.w = v[3];
    *(float4*)p = t;
}
__device__ __forceinline__ float sigmoidf_(float x) { return 1.0f / (1.0f + __expf(-x)); }
__device__ __forceinline__ float softplusf_(float x) { return (x > 20.0f) ? x : log1pf(__expf(x)); }

// ---------------------------------------------------------------- embed
__global__ void embed_kernel(const int* __restrict__ x_idx, const float* __restrict__ emb,
                             float* __restrict__ x) {
    int i = blockIdx.x * blockDim.x + threadIdx.x;           // over N*128
    if (i >= NN * DD) return;
    int n = i >> 7, c = i & 127;
    x[i] = emb[x_idx[n] * DD + c];
}

// ---------------------------------------------------------------- f32 -> f16 convert (x)
__global__ void cvt_x_kernel(const float* __restrict__ x, _Float16* __restrict__ xh) {
    int i = blockIdx.x * blockDim.x + threadIdx.x;
    if (i < NN * DD) xh[i] = (_Float16)x[i];
}

// W blocks -> f16, transposed: whT[mat][col c][k], mat 0/1 = Wf rows [0:128)/[128:256),
// mat 2/3 = Ws rows [0:128)/[128:256).
__global__ void cvt_w_kernel(const float* __restrict__ Wf_l, const float* __restrict__ Ws_l,
                             _Float16* __restrict__ whT) {
    int i = blockIdx.x * blockDim.x + threadIdx.x;           // over 4*128*128
    if (i >= 4 * DD * DD) return;
    int mat = i >> 14;
    int c = (i >> 7) & 127;
    int k = i & 127;
    const float* src = (mat < 2) ? Wf_l : Ws_l;
    int rowoff = (mat & 1) ? DD : 0;
    whT[i] = (_Float16)src[(rowoff + k) * DD + c];
}

// ---------------------------------------------------------------- WMMA node projections
// P[mat][n][c] = sum_k x[n][k] * W_mat[k][c],   4 mats selected by blockIdx.y.
// One wave computes one 16x16 C tile; block = 8 waves = full 128-wide output strip.
__global__ void gemm_node_proj(const _Float16* __restrict__ xh,
                               const _Float16* __restrict__ whT,   // [4][128][128] (col-major K)
                               float* __restrict__ P)              // [4][NN][128]
{
    const int w    = threadIdx.x >> 5;      // n-tile 0..7
    const int lane = threadIdx.x & 31;
    const int mt   = blockIdx.x;            // m-tile 0..3124 (NN = 3125*16 exactly)
    const int mat  = blockIdx.y;            // which W / which P

    const _Float16* W  = whT + (size_t)mat * DD * DD;
    float* out         = P   + (size_t)mat * NN * DD;

    const int m        = lane & 15;
    const int half_hi  = lane >> 4;         // 0: lanes 0-15, 1: lanes 16-31
    const int row      = mt * 16 + m;       // A row (node index)
    const int col0     = w * 16;
    const int n        = lane & 15;         // B/C column within tile

    const _Float16* arow = xh + (size_t)row * DD;
    const _Float16* bcol = W  + (size_t)(col0 + n) * DD;

    v8f c = {};
    #pragma unroll
    for (int kb = 0; kb < 4; ++kb) {
        // A (16x32 f16): lane half picks k-octet; halves [0:8)=k0..k0+7, [8:16)=k0+16..k0+23
        int k0 = kb * 32 + half_hi * 8;
        v8h alo = *(const v8h*)(arow + k0);
        v8h ahi = *(const v8h*)(arow + k0 + 16);
        v16h a;
        #pragma unroll
        for (int i = 0; i < 8; ++i) { a[i] = alo[i]; a[i + 8] = ahi[i]; }

        // B (32x16 f16): same k order as A: [0:8)=k0..k0+7, [8:16)=k0+16..k0+23
        int kB = kb * 32 + half_hi * 8;
        v8h blo = *(const v8h*)(bcol + kB);
        v8h bhi = *(const v8h*)(bcol + kB + 16);
        v16h b;
        #pragma unroll
        for (int i = 0; i < 8; ++i) { b[i] = blo[i]; b[i + 8] = bhi[i]; }

        c = __builtin_amdgcn_wmma_f32_16x16x32_f16(false, a, false, b, (short)0, c, false, false);
    }

    // C layout: VGPR r -> M = r (lanes 0-15) / r+8 (lanes 16-31); N = lane&15
    const int rbase = mt * 16 + half_hi * 8;
    #pragma unroll
    for (int r = 0; r < 8; ++r)
        out[(size_t)(rbase + r) * DD + col0 + n] = c[r];
}

// ---------------------------------------------------------------- copy x -> xnew (float4)
__global__ void copy4_kernel(const float* __restrict__ src, float* __restrict__ dst) {
    int i = blockIdx.x * blockDim.x + threadIdx.x;           // over N*128/4
    if (i < NN * DD / 4) ((float4*)dst)[i] = ((const float4*)src)[i];
}

// ---------------------------------------------------------------- zero acc
__global__ void zero_kernel(float* __restrict__ p, int n) {
    int i = blockIdx.x * blockDim.x + threadIdx.x;
    if (i < n) p[i] = 0.0f;
}

// ---------------------------------------------------------------- edge message passing
// m_e = sigmoid(Pfd[dst]+Pfs[src]+ea@Wfe+bf) * softplus(Psd[dst]+Pss[src]+ea@Wse+bs)
// xnew[dst] += m_e   (fp32 atomics). One wave per edge, 8 edges per wave, 64 per block.
__global__ void edge_mp_kernel(const int* __restrict__ srcI, const int* __restrict__ dstI,
                               const float* __restrict__ ea,
                               const float* __restrict__ Wf_l, const float* __restrict__ bf_l,
                               const float* __restrict__ Ws_l, const float* __restrict__ bs_l,
                               const float* __restrict__ Pfd, const float* __restrict__ Pfs,
                               const float* __restrict__ Psd, const float* __restrict__ Pss,
                               float* __restrict__ xnew)
{
    __shared__ float WfeS[EDD * DD];
    __shared__ float WseS[EDD * DD];
    __shared__ float bfS[DD], bsS[DD];
    for (int j = threadIdx.x; j < EDD * DD; j += blockDim.x) {
        int r = j >> 7, cc = j & 127;
        WfeS[j] = Wf_l[(2 * DD + r) * DD + cc];
        WseS[j] = Ws_l[(2 * DD + r) * DD + cc];
    }
    for (int j = threadIdx.x; j < DD; j += blockDim.x) { bfS[j] = bf_l[j]; bsS[j] = bs_l[j]; }
    __syncthreads();

    const int wave = threadIdx.x >> 5;
    const int lane = threadIdx.x & 31;
    const int c0 = lane * 4;

    for (int it = 0; it < 8; ++it) {
        int e = blockIdx.x * 64 + wave * 8 + it;
        if (e >= EE) return;
        int s = srcI[e], d = dstI[e];
        float eav = ea[(size_t)e * EDD + (lane & 15)];     // lanes k and k+16 hold ea[k]

        float fv[4], sv[4], t0[4], t1[4];
        load4(Pfd + (size_t)d * DD + c0, fv);
        load4(Pfs + (size_t)s * DD + c0, t0);
        load4(Psd + (size_t)d * DD + c0, sv);
        load4(Pss + (size_t)s * DD + c0, t1);
        #pragma unroll
        for (int j = 0; j < 4; ++j) {
            fv[j] += t0[j] + bfS[c0 + j];
            sv[j] += t1[j] + bsS[c0 + j];
        }
        #pragma unroll
        for (int k = 0; k < EDD; ++k) {
            float a = __shfl(eav, k, 32);
            float wf[4], wsv[4];
            load4(&WfeS[k * DD + c0], wf);
            load4(&WseS[k * DD + c0], wsv);
            #pragma unroll
            for (int j = 0; j < 4; ++j) { fv[j] += a * wf[j]; sv[j] += a * wsv[j]; }
        }
        #pragma unroll
        for (int j = 0; j < 4; ++j) {
            float m = sigmoidf_(fv[j]) * softplusf_(sv[j]);
            atomicAdd(&xnew[(size_t)d * DD + c0 + j], m);
        }
    }
}

// ---------------------------------------------------------------- relu + per-node dots
// x = relu(xnew); t_rel[n] = x[n].wrel; sbuf[n] = brel + x[n].wroot
__global__ void relu_score_kernel(float* __restrict__ x, const float* __restrict__ xnew,
                                  const float* __restrict__ Wrel_l, const float* __restrict__ brel_l,
                                  const float* __restrict__ Wroot_l,
                                  float* __restrict__ t_rel, float* __restrict__ sbuf)
{
    int wave = threadIdx.x >> 5, lane = threadIdx.x & 31;
    int node = blockIdx.x * 8 + wave;                       // grid = NN/8 = 6250
    if (node >= NN) return;
    int c0 = lane * 4;
    float v[4], wr[4], wo[4];
    load4(xnew + (size_t)node * DD + c0, v);
    load4(Wrel_l + c0, wr);
    load4(Wroot_l + c0, wo);
    float dr = 0.0f, dro = 0.0f;
    #pragma unroll
    for (int j = 0; j < 4; ++j) {
        v[j] = fmaxf(v[j], 0.0f);
        dr  += v[j] * wr[j];
        dro += v[j] * wo[j];
    }
    store4(x + (size_t)node * DD + c0, v);
    #pragma unroll
    for (int off = 16; off >= 1; off >>= 1) {
        dr  += __shfl_xor(dr,  off, 32);
        dro += __shfl_xor(dro, off, 32);
    }
    if (lane == 0) { t_rel[node] = dr; sbuf[node] = brel_l[0] + dro; }
}

// ---------------------------------------------------------------- scalar rel scatter
__global__ void scatter_rel_kernel(const int* __restrict__ srcI, const int* __restrict__ dstI,
                                   const float* __restrict__ t_rel, float* __restrict__ sbuf)
{
    int e = blockIdx.x * blockDim.x + threadIdx.x;
    if (e < EE) atomicAdd(&sbuf[dstI[e]], t_rel[srcI[e]]);
}

// ---------------------------------------------------------------- per-graph softmax pool
// Graph g owns nodes [ceil(g*N/B), ceil((g+1)*N/B))  (batch = (n*B)//N, sorted).
// x *= score; acc[g][0:128] += chan-max; acc[g][128:256] += chan-mean.
__global__ void sag_pool_kernel(float* __restrict__ x, const float* __restrict__ sbuf,
                                float* __restrict__ acc)
{
    int g = blockIdx.x, t = threadIdx.x;                    // 128 threads
    long start = ((long)g * NN + BB - 1) / BB;
    long end   = ((long)(g + 1) * NN + BB - 1) / BB;
    int cnt = (int)(end - start);                           // <= 196

    __shared__ float red[128];
    __shared__ float scoreS[256];

    float mx = -1e30f;
    for (int n = t; n < cnt; n += 128) mx = fmaxf(mx, sbuf[start + n]);
    red[t] = mx; __syncthreads();
    for (int o = 64; o >= 1; o >>= 1) { if (t < o) red[t] = fmaxf(red[t], red[t + o]); __syncthreads(); }
    float smax = red[0]; __syncthreads();

    float sm = 0.0f;
    for (int n = t; n < cnt; n += 128) {
        float e = __expf(sbuf[start + n] - smax);
        scoreS[n] = e;
        sm += e;
    }
    red[t] = sm; __syncthreads();
    for (int o = 64; o >= 1; o >>= 1) { if (t < o) red[t] += red[t + o]; __syncthreads(); }
    float inv = 1.0f / red[0];

    float chmax = -1e30f, chsum = 0.0f;
    for (int n = 0; n < cnt; ++n) {
        float sc = scoreS[n] * inv;
        size_t idx = (size_t)(start + n) * DD + t;
        float v = x[idx] * sc;
        x[idx] = v;
        chmax = fmaxf(chmax, v);
        chsum += v;
    }
    acc[g * 2 * DD + t]      += chmax;
    acc[g * 2 * DD + DD + t] += chsum / (float)cnt;
}

// ---------------------------------------------------------------- tiny MLP layer
__global__ void mlp_kernel(const float* __restrict__ in, const float* __restrict__ W,
                           const float* __restrict__ bias, float* __restrict__ out,
                           int K, int relu)
{
    __shared__ float inS[256];
    int b = blockIdx.x, t = threadIdx.x;                    // 128 threads
    for (int j = t; j < K; j += 128) inS[j] = in[b * K + j];
    __syncthreads();
    float a = bias[t];
    for (int k = 0; k < K; ++k) a += inS[k] * W[k * DD + t];
    if (relu) a = fmaxf(a, 0.0f);
    out[b * DD + t] = a;
}

// ---------------------------------------------------------------- log_softmax rows
__global__ void logsoftmax_kernel(const float* __restrict__ h, float* __restrict__ out)
{
    __shared__ float red[128];
    __shared__ float mxs, sms;
    int b = blockIdx.x, t = threadIdx.x;
    float v = h[b * DD + t];
    red[t] = v; __syncthreads();
    for (int o = 64; o >= 1; o >>= 1) { if (t < o) red[t] = fmaxf(red[t], red[t + o]); __syncthreads(); }
    if (t == 0) mxs = red[0];
    __syncthreads();
    float e = __expf(v - mxs);
    red[t] = e; __syncthreads();
    for (int o = 64; o >= 1; o >>= 1) { if (t < o) red[t] += red[t + o]; __syncthreads(); }
    if (t == 0) sms = red[0];
    __syncthreads();
    out[b * DD + t] = v - mxs - logf(sms);
}

// ================================================================ host
extern "C" void kernel_launch(void* const* d_in, const int* in_sizes, int n_in,
                              void* d_out, int out_size, void* d_ws, size_t ws_size,
                              hipStream_t stream) {
    (void)in_sizes; (void)n_in; (void)out_size; (void)ws_size;
    const int*   x_idx = (const int*)  d_in[0];
    const int*   eidx  = (const int*)  d_in[1];   // [2, E]
    const float* ea    = (const float*)d_in[2];   // [E, 16]
    const float* emb   = (const float*)d_in[4];   // [100, 128]
    const float* Wf    = (const float*)d_in[5];   // [L, 272, 128]
    const float* bf    = (const float*)d_in[6];   // [L, 128]
    const float* Ws    = (const float*)d_in[7];
    const float* bs    = (const float*)d_in[8];
    const float* Wrel  = (const float*)d_in[9];   // [L, 128, 1]
    const float* brel  = (const float*)d_in[10];  // [L, 1]
    const float* Wroot = (const float*)d_in[11];  // [L, 128, 1]
    const float* W1    = (const float*)d_in[12];  // [256, 128]
    const float* b1    = (const float*)d_in[13];
    const float* W2    = (const float*)d_in[14];  // [128, 128]
    const float* b2    = (const float*)d_in[15];
    const float* W3    = (const float*)d_in[16];  // [128, 128]
    const float* b3    = (const float*)d_in[17];
    float* out = (float*)d_out;

    const int* srcI = eidx;
    const int* dstI = eidx + EE;

    // ---- workspace carve (256B aligned slabs)
    char* p = (char*)d_ws;
    auto carve = [&](size_t bytes) {
        void* r = (void*)p;
        p += (bytes + 255) & ~(size_t)255;
        return r;
    };
    float*    x    = (float*)   carve((size_t)NN * DD * 4);
    float*    xnew = (float*)   carve((size_t)NN * DD * 4);
    _Float16* xh   = (_Float16*)carve((size_t)NN * DD * 2);
    float*    P    = (float*)   carve((size_t)4 * NN * DD * 4);  // Pfd,Pfs,Psd,Pss
    _Float16* whT  = (_Float16*)carve((size_t)4 * DD * DD * 2);
    float*    trel = (float*)   carve((size_t)NN * 4);
    float*    sbuf = (float*)   carve((size_t)NN * 4);
    float*    acc  = (float*)   carve((size_t)BB * 2 * DD * 4);
    float*    h1   = (float*)   carve((size_t)BB * DD * 4);
    float*    h2   = (float*)   carve((size_t)BB * DD * 4);

    const size_t Pstride = (size_t)NN * DD;

    // ---- init
    embed_kernel<<<(NN * DD + 255) / 256, 256, 0, stream>>>(x_idx, emb, x);
    zero_kernel<<<(BB * 2 * DD + 255) / 256, 256, 0, stream>>>(acc, BB * 2 * DD);

    // ---- 4 conv+pool stages
    for (int l = 0; l < LL; ++l) {
        const float* Wf_l = Wf + (size_t)l * 272 * DD;
        const float* Ws_l = Ws + (size_t)l * 272 * DD;

        cvt_x_kernel<<<(NN * DD + 255) / 256, 256, 0, stream>>>(x, xh);
        cvt_w_kernel<<<(4 * DD * DD + 255) / 256, 256, 0, stream>>>(Wf_l, Ws_l, whT);

        gemm_node_proj<<<dim3(NN / 16, 4), 256, 0, stream>>>(xh, whT, P);

        copy4_kernel<<<(NN * DD / 4 + 255) / 256, 256, 0, stream>>>(x, xnew);

        edge_mp_kernel<<<(EE + 63) / 64, 256, 0, stream>>>(
            srcI, dstI, ea, Wf_l, bf + (size_t)l * DD, Ws_l, bs + (size_t)l * DD,
            P + 0 * Pstride, P + 1 * Pstride, P + 2 * Pstride, P + 3 * Pstride, xnew);

        relu_score_kernel<<<NN / 8, 256, 0, stream>>>(
            x, xnew, Wrel + (size_t)l * DD, brel + l, Wroot + (size_t)l * DD, trel, sbuf);

        scatter_rel_kernel<<<(EE + 255) / 256, 256, 0, stream>>>(srcI, dstI, trel, sbuf);

        sag_pool_kernel<<<BB, 128, 0, stream>>>(x, sbuf, acc);
    }

    // ---- head MLP + log_softmax
    mlp_kernel<<<BB, 128, 0, stream>>>(acc, W1, b1, h1, 2 * DD, 1);
    mlp_kernel<<<BB, 128, 0, stream>>>(h1, W2, b2, h2, DD, 1);
    mlp_kernel<<<BB, 128, 0, stream>>>(h2, W3, b3, h1, DD, 0);   // reuse h1 as logits
    logsoftmax_kernel<<<BB, 128, 0, stream>>>(h1, out);
}
